// FourierDecoder_86285892977133
// MI455X (gfx1250) — matrix operand order repaired and checked by reading, not compiled, not run on hardware
//
#include <hip/hip_runtime.h>
#include <type_traits>

typedef __attribute__((ext_vector_type(16))) _Float16 v16h;
typedef __attribute__((ext_vector_type(8)))  float    v8f;
typedef __attribute__((ext_vector_type(4)))  unsigned v4u;
typedef __attribute__((ext_vector_type(8)))  unsigned v8u;

#define DI __device__ __forceinline__

union H8 { uint4 u; _Float16 h[8]; };

// A-fragment (16x32 f16): lane m = lane&15, kbase = (lane&16)?16:0,
// elements 0..15 -> K=kbase+0..15 (matches B fragment k order)
DI v16h ld_frag_a(const _Float16* s) {
  union { uint4 u[2]; v16h v; } r;
  r.u[0] = *(const uint4*)(s);
  r.u[1] = *(const uint4*)(s + 8);
  return r.v;
}

// B-fragment (32x16 f16) from K-pair-interleaved LDS tile:
// element e -> K=kbase+e; pairs (2e,2e+1) are contiguous b32 at group row e.
DI v16h ld_frag_b2(const _Float16* s, int pitch) {
  union { unsigned u[8]; v16h v; } r;
#pragma unroll
  for (int e = 0; e < 8; ++e) r.u[e] = *(const unsigned*)(s + (long)e * pitch);
  return r.v;
}

DI float silu_f(float x) { return x / (1.f + __expf(-x)); }

// ---- Tensor Data Mover: one 2D tile Global->LDS, HW row padding to BKP pitch ----
// D# group0: [1:0]=count=1, [63:32]=lds_addr, [120:64]=global_addr, [127:126]=type=2
// D# group1: data_size=2B, pad_enable, pad_interval=16 DW (=32 halves),
//            pad_amount=4 DW (=8 halves -> 40-half pitch), dims/strides below.
DI void tdm_load_tile(unsigned lds_byte, const void* gsrc,
                      unsigned tile_rows, unsigned tile_cols,
                      unsigned tensor_cols, unsigned tensor_rows,
                      unsigned long row_stride_elems) {
  unsigned long ga = (unsigned long)gsrc;
  v4u g0;
  g0[0] = 1u;                                               // count=1, user mode
  g0[1] = lds_byte;                                         // LDS byte address
  g0[2] = (unsigned)ga;                                     // global addr [31:0]
  g0[3] = (unsigned)((ga >> 32) & 0x01FFFFFFul) | (2u << 30); // [56:32] | type=2
  v8u g1;
  g1[0] = (1u << 16)   // data_size = 1 -> 2 bytes
        | (1u << 20)   // pad_enable (load only)
        | (3u << 22)   // pad_interval code 3 -> every 16 DWORDs
        | (3u << 25);  // pad_amount  code 3 -> 4 DWORDs
  g1[1] = (tensor_cols & 0xFFFFu) << 16;                    // tensor_dim0[15:0] @bit48
  g1[2] = (tensor_cols >> 16) | ((tensor_rows & 0xFFFFu) << 16);
  g1[3] = (tensor_rows >> 16) | (tile_cols << 16);          // tile_dim0 @bit112
  g1[4] = tile_rows;                                        // tile_dim1 (tile_dim2=0)
  g1[5] = (unsigned)row_stride_elems;                       // dim0_stride[31:0] @bit160
  g1[6] = (unsigned)((row_stride_elems >> 32) & 0xFFFFul);  // dim0_stride[47:32]
  g1[7] = 0u;
  asm volatile("tensor_load_to_lds %0, %1" :: "s"(g0), "s"(g1) : "memory");
}

enum { EPI_BIAS = 0, EPI_FWD = 1, EPI_SPECTRAL = 2, EPI_INV_SILU = 3, EPI_MLP1 = 4 };

struct GArgs {
  const void* A; const void* B;
  int lda, ldb;
  long sAz, sBz;          // z (batch / mode) strides in elements
  int M, N, K;
  const float* bias;
  const _Float16* res;    // residual (f16) for EPI_INV_SILU
  void* C;
  int ldc; long sCz;
  int Cs;                 // split size (Cin for FWD, Cout for SPECTRAL)
};

template<int BM, int BN, int BK, bool AF32, bool BF32, int EPI>
__global__ __launch_bounds__(256) void gemm_wmma(GArgs g) {
  static_assert(BN == 128 && BK == 32, "tiling fixed");
  constexpr int BKP = BK + 8;           // A row pitch (halves): 80B rows, conflict-free
  constexpr int BNI = 2 * BN + 16;      // B k-pair group pitch (halves): 544B rows
  constexpr int ASZ = BM * BKP;
  constexpr int BSZ = (BK / 2) * BNI;
  __shared__ __attribute__((aligned(16))) _Float16 As[2 * ASZ];
  __shared__ __attribute__((aligned(16))) _Float16 Bs[2 * BSZ];

  const int tid  = threadIdx.x;
  const int lane = tid & 31;
  const int wid  = tid >> 5;
  const int wr = wid >> 2, wc = wid & 3;          // 2x4 wave grid
  constexpr int WM = BM / 2, WN = BN / 4;
  constexpr int TM = WM / 16, TN = WN / 16;

  const int z  = blockIdx.z;
  const int m0 = blockIdx.y * BM;
  const int n0 = blockIdx.x * BN;

  using AT = typename std::conditional<AF32, float, _Float16>::type;
  using BT = typename std::conditional<BF32, float, _Float16>::type;
  const AT* Ap = (const AT*)g.A + (long)z * g.sAz;
  const BT* Bp = (const BT*)g.B + (long)z * g.sBz;

  constexpr int SEGA = BK / 8;              // 4 chunks per A row
  constexpr int CHA  = BM * BK / 8;         // 16B chunks in A tile

  // ---- A tile staging: TDM (f16) or synchronous convert (f32 source) ----
  auto stageA = [&](int buf, int k0) {
    if (AF32) {
      for (int c = tid; c < CHA; c += 256) {
        int r = c / SEGA, s8 = (c % SEGA) * 8;
        long go = (long)(m0 + r) * g.lda + k0 + s8;
        const float* src = (const float*)Ap + go;
        float4 f0 = *(const float4*)(src);
        float4 f1 = *(const float4*)(src + 4);
        H8 hh;
        hh.h[0]=(_Float16)f0.x; hh.h[1]=(_Float16)f0.y;
        hh.h[2]=(_Float16)f0.z; hh.h[3]=(_Float16)f0.w;
        hh.h[4]=(_Float16)f1.x; hh.h[5]=(_Float16)f1.y;
        hh.h[6]=(_Float16)f1.z; hh.h[7]=(_Float16)f1.w;
        *(uint4*)&As[buf * ASZ + r * BKP + s8] = hh.u;
      }
    } else {
      // single TDM DMA per tile, issued by wave 0 only (scalar branch: TDM ignores EXEC)
      if (__builtin_amdgcn_readfirstlane(tid) == 0) {
        unsigned lds = (unsigned)(unsigned long)&As[buf * ASZ];
        const _Float16* gp = (const _Float16*)Ap + (long)m0 * g.lda + k0;
        tdm_load_tile(lds, gp, (unsigned)BM, (unsigned)BK,
                      (unsigned)g.lda, (unsigned)BM, (unsigned long)g.lda);
      }
    }
  };

  // ---- B tile: software-pipelined; one 16-half chunk (k-pair x 8 cols) per thread ----
  const int bK2 = tid >> 4;          // k-pair group 0..15
  const int bN8 = (tid & 15) * 8;    // col segment
  H8 bra, brb;                       // rows 2*bK2 and 2*bK2+1
  auto loadB = [&](int k0) {
    long go0 = (long)(k0 + 2 * bK2) * g.ldb + n0 + bN8;
    long go1 = go0 + g.ldb;
    if (BF32) {
      const float* s0 = (const float*)Bp + go0;
      const float* s1 = (const float*)Bp + go1;
      float4 a0 = *(const float4*)(s0), a1 = *(const float4*)(s0 + 4);
      float4 b0 = *(const float4*)(s1), b1 = *(const float4*)(s1 + 4);
      bra.h[0]=(_Float16)a0.x; bra.h[1]=(_Float16)a0.y;
      bra.h[2]=(_Float16)a0.z; bra.h[3]=(_Float16)a0.w;
      bra.h[4]=(_Float16)a1.x; bra.h[5]=(_Float16)a1.y;
      bra.h[6]=(_Float16)a1.z; bra.h[7]=(_Float16)a1.w;
      brb.h[0]=(_Float16)b0.x; brb.h[1]=(_Float16)b0.y;
      brb.h[2]=(_Float16)b0.z; brb.h[3]=(_Float16)b0.w;
      brb.h[4]=(_Float16)b1.x; brb.h[5]=(_Float16)b1.y;
      brb.h[6]=(_Float16)b1.z; brb.h[7]=(_Float16)b1.w;
    } else {
      bra.u = *(const uint4*)((const _Float16*)Bp + go0);
      brb.u = *(const uint4*)((const _Float16*)Bp + go1);
    }
  };
  auto storeB = [&](int buf) {
    union { uint4 u[2]; _Float16 h[16]; } o;
#pragma unroll
    for (int i = 0; i < 8; ++i) { o.h[2 * i] = bra.h[i]; o.h[2 * i + 1] = brb.h[i]; }
    uint4* d = (uint4*)&Bs[buf * BSZ + bK2 * BNI + bN8 * 2];
    d[0] = o.u[0]; d[1] = o.u[1];
  };

  v8f acc[TM][TN] = {};
  const int lm  = lane & 15;
  const int kAo = (lane & 16) ? 16 : 0;  // A k-offset per half-wave
  const int kBg = (lane & 16) ? 8 : 0;   // B k-pair group offset (= 16/2)
  const int NT  = g.K / BK;

  // prologue: tile 0 into buffer 0
  loadB(0);
  stageA(0, 0);
  storeB(0);

  for (int t = 0; t < NT; ++t) {
    const int cb = t & 1, nb = cb ^ 1;
    const bool more = (t + 1 < NT);
    if (more) { loadB((t + 1) * BK); stageA(nb, (t + 1) * BK); }
    if (!AF32) {
      // issuing wave: tile t done when <= 1 (tile t+1) remains; others: counter==0, no-op
      if (more) __builtin_amdgcn_s_wait_tensorcnt((short)1);
      else      __builtin_amdgcn_s_wait_tensorcnt((short)0);
    }
    __syncthreads();

    const _Float16* Ab = &As[cb * ASZ];
    const _Float16* Bb = &Bs[cb * BSZ];
    v16h bf[TN];
#pragma unroll
    for (int tj = 0; tj < TN; ++tj)
      bf[tj] = ld_frag_b2(&Bb[kBg * BNI + (wc * WN + tj * 16 + lm) * 2], BNI);
#pragma unroll
    for (int ti = 0; ti < TM; ++ti) {
      v16h af = ld_frag_a(&Ab[(wr * WM + ti * 16 + lm) * BKP + kAo]);
#pragma unroll
      for (int tj = 0; tj < TN; ++tj)
        acc[ti][tj] = __builtin_amdgcn_wmma_f32_16x16x32_f16(
            false, af, false, bf[tj], (short)0, acc[ti][tj], false, false);
    }
    if (more) storeB(nb);      // ds_store hidden behind this tile's WMMAs
    __syncthreads();
  }

  // ---- epilogue: C/D layout: VGPR r -> row r + ((lane&16)?8:0), col = lane&15 ----
  const int ml = (lane & 16) ? 8 : 0;
  const int nl = lane & 15;
#pragma unroll
  for (int ti = 0; ti < TM; ++ti) {
#pragma unroll
    for (int tj = 0; tj < TN; ++tj) {
      const int gmb = m0 + wr * WM + ti * 16 + ml;
      const int gn  = n0 + wc * WN + tj * 16 + nl;
#pragma unroll
      for (int r = 0; r < 8; ++r) {
        float v = acc[ti][tj][r];
        int gm = gmb + r;
        if (EPI == EPI_BIAS) {
          ((_Float16*)g.C)[(long)gm * g.ldc + gn] = (_Float16)(v + g.bias[gn]);
        } else if (EPI == EPI_FWD) {
          // rows 0..63 = real -> [k][0:Cin], rows 64..127 = imag -> [k][Cin:2Cin]
          _Float16* o = (_Float16*)g.C + (long)z * g.sCz;
          long idx = (gm < 64) ? (long)gm * g.ldc + gn
                               : (long)(gm - 64) * g.ldc + g.Cs + gn;
          o[idx] = (_Float16)v;
        } else if (EPI == EPI_SPECTRAL) {
          // cols 0..Cout-1 = Or -> row z ; cols Cout.. = Oi -> row 64+z
          _Float16* o = (_Float16*)g.C;
          long base = (long)gm * (128L * g.Cs);
          long idx = (gn < g.Cs) ? base + (long)z * g.Cs + gn
                                 : base + (long)(64 + z) * g.Cs + (gn - g.Cs);
          o[idx] = (_Float16)v;
        } else if (EPI == EPI_INV_SILU) {
          long idx = (long)z * g.sCz + (long)gm * g.ldc + gn;
          float s = v + (float)g.res[idx];
          ((_Float16*)g.C)[idx] = (_Float16)silu_f(s);
        } else { // EPI_MLP1
          ((_Float16*)g.C)[(long)gm * g.ldc + gn] =
              (_Float16)silu_f(v + g.bias[gn]);
        }
      }
    }
  }
}

// Fwd basis Fb[128][2048]: rows 0..63 = cos, 64..127 = -sin.
// Inv basis Gi[2048][128]: cols 0..63 scale*cos, 64..127 scale*(-sin); irfft scaling folded in.
__global__ void k_basis(_Float16* Fb, _Float16* Gi) {
  int idx = blockIdx.x * 256 + threadIdx.x;
  if (idx >= 64 * 2048) return;
  int k = idx >> 11, gg = idx & 2047;
  float th = 3.0679615757e-3f * (float)((k * gg) & 2047); // 2*pi/2048, exact periodic arg
  float s = __sinf(th), c = __cosf(th);
  Fb[k * 2048 + gg]        = (_Float16)c;
  Fb[(64 + k) * 2048 + gg] = (_Float16)(-s);
  float sc = (k == 0) ? (1.f / 2048.f) : (2.f / 2048.f);
  Gi[gg * 128 + k]      = (_Float16)(c * sc);
  Gi[gg * 128 + 64 + k] = (_Float16)((k == 0) ? 0.f : (-s * sc));
}

// Complex w[Cin][Cout][64] (interleaved) -> per-mode real B: [[Wr, Wi], [-Wi, Wr]]
__global__ void k_wspec(const float* w, _Float16* out, int Cin, int Cout) {
  long idx = (long)blockIdx.x * 256 + threadIdx.x;
  long total = 64L * Cin * Cout;
  if (idx >= total) return;
  int o = (int)(idx % Cout);
  long t = idx / Cout;
  int i = (int)(t % Cin);
  int k = (int)(t / Cin);
  long src = ((long)(i * Cout + o) * 64 + k) * 2;
  float re = w[src], im = w[src + 1];
  int ld = 2 * Cout;
  _Float16* ob = out + (long)k * (2 * Cin) * ld;
  ob[(long)i * ld + o]                = (_Float16)re;
  ob[(long)i * ld + Cout + o]         = (_Float16)im;
  ob[(long)(Cin + i) * ld + o]        = (_Float16)(-im);
  ob[(long)(Cin + i) * ld + Cout + o] = (_Float16)re;
}

__global__ void k_cvt(const float* src, _Float16* dst, long n) {
  long i = (long)blockIdx.x * 256 + threadIdx.x;
  if (i < n) dst[i] = (_Float16)src[i];
}

// out[row] = silu-ed hidden row . f2 + f2b   (one wave32 per row)
__global__ void k_dot(const _Float16* t, const float* f2, const float* f2b,
                      float* out, long rows) {
  long row = (long)blockIdx.x * 8 + (threadIdx.x >> 5);
  if (row >= rows) return;
  int lane = threadIdx.x & 31;
  const _Float16* p = t + row * 128 + lane * 4;
  float acc = 0.f;
#pragma unroll
  for (int j = 0; j < 4; ++j) acc += (float)p[j] * f2[lane * 4 + j];
#pragma unroll
  for (int off = 16; off > 0; off >>= 1) acc += __shfl_down(acc, off, 32);
  if (lane == 0) out[row] = acc + f2b[0];
}

extern "C" void kernel_launch(void* const* d_in, const int* in_sizes, int n_in,
                              void* d_out, int out_size, void* d_ws, size_t ws_size,
                              hipStream_t stream) {
  const float* x    = (const float*)d_in[0];
  const float* w0   = (const float*)d_in[1];   // complex64 interleaved
  const float* l0W  = (const float*)d_in[2];
  const float* l0b  = (const float*)d_in[3];
  const float* wsC  = (const float*)d_in[4];   // [3][128][128][64] complex
  const float* lsW  = (const float*)d_in[5];
  const float* lsb  = (const float*)d_in[6];
  const float* f1W  = (const float*)d_in[7];
  const float* f1b  = (const float*)d_in[8];
  const float* f2W  = (const float*)d_in[9];
  const float* f2b  = (const float*)d_in[10];
  (void)in_sizes; (void)n_in; (void)out_size; (void)ws_size;

  const int B = 64, G = 2048, D0 = 256, W = 128, MODES = 64;
  const long BG = (long)B * G;

  char* p = (char*)d_ws;
  auto alloc = [&](size_t bytes) {
    char* r = p; p += (bytes + 255) & ~(size_t)255; return r;
  };
  _Float16* h16   = (_Float16*)alloc((size_t)BG * W * 2);            // activations (f16)
  _Float16* res16 = (_Float16*)alloc((size_t)BG * W * 2);            // residual / MLP temp
  _Float16* xft   = (_Float16*)alloc((size_t)B * MODES * 2 * D0 * 2);// [b][k][2Cin]
  _Float16* obuf  = (_Float16*)alloc((size_t)B * 2 * MODES * W * 2); // [b][Or0..63,Oi0..63][Cout]
  _Float16* Fb    = (_Float16*)alloc((size_t)128 * G * 2);
  _Float16* Gi    = (_Float16*)alloc((size_t)G * 128 * 2);
  _Float16* wspec = (_Float16*)alloc((size_t)MODES * 2 * D0 * 2 * W * 2);
  _Float16* lw16  = (_Float16*)alloc((size_t)D0 * W * 2);
  _Float16* f116  = (_Float16*)alloc((size_t)W * W * 2);
  _Float16* t16   = res16; // reuse after last layer

  k_basis<<<dim3((64 * G + 255) / 256), 256, 0, stream>>>(Fb, Gi);

  GArgs a;

  for (int layer = 0; layer < 4; ++layer) {
    const int Cin = (layer == 0) ? D0 : W;
    const float* lWf = (layer == 0) ? l0W : lsW + (size_t)(layer - 1) * W * W;
    const float* lbf = (layer == 0) ? l0b : lsb + (size_t)(layer - 1) * W;
    const float* wf  = (layer == 0) ? w0  : wsC + (size_t)(layer - 1) * W * W * MODES * 2;

    k_cvt<<<dim3((Cin * W + 255) / 256), 256, 0, stream>>>(lWf, lw16, (long)Cin * W);
    k_wspec<<<dim3((int)((64L * Cin * W + 255) / 256)), 256, 0, stream>>>(wf, wspec, Cin, W);

    // residual: res = act @ lW + lb   [BG x Cin] @ [Cin x W]
    a = {};
    a.B = lw16; a.ldb = W;
    a.M = (int)BG; a.N = W; a.K = Cin;
    a.bias = lbf; a.C = res16; a.ldc = W;
    if (layer == 0) {
      a.A = x; a.lda = D0;
      gemm_wmma<128,128,32,true ,false,EPI_BIAS>
          <<<dim3(1, (int)(BG / 128), 1), 256, 0, stream>>>(a);
    } else {
      a.A = h16; a.lda = W;
      gemm_wmma<128,128,32,false,false,EPI_BIAS>
          <<<dim3(1, (int)(BG / 128), 1), 256, 0, stream>>>(a);
    }

    // forward DFT: Xft_b = Fb[128 x 2048] @ act_b[2048 x Cin]
    a = {};
    a.A = Fb; a.lda = G; a.sAz = 0;
    a.M = 128; a.N = Cin; a.K = G;
    a.C = xft; a.ldc = 2 * Cin; a.sCz = (long)MODES * 2 * Cin; a.Cs = Cin;
    if (layer == 0) {
      a.B = x; a.ldb = D0; a.sBz = (long)G * D0;
      gemm_wmma<128,128,32,false,true ,EPI_FWD>
          <<<dim3(Cin / 128, 1, B), 256, 0, stream>>>(a);
    } else {
      a.B = h16; a.ldb = W; a.sBz = (long)G * W;
      gemm_wmma<128,128,32,false,false,EPI_FWD>
          <<<dim3(Cin / 128, 1, B), 256, 0, stream>>>(a);
    }

    // spectral: per mode k: [B x 2Cin] @ [2Cin x 2W]
    a = {};
    a.A = xft;   a.lda = MODES * 2 * Cin; a.sAz = 2 * Cin;
    a.B = wspec; a.ldb = 2 * W;           a.sBz = (long)(2 * Cin) * (2 * W);
    a.M = B; a.N = 2 * W; a.K = 2 * Cin;
    a.C = obuf; a.Cs = W;
    gemm_wmma<64,128,32,false,false,EPI_SPECTRAL>
        <<<dim3((2 * W) / 128, 1, MODES), 256, 0, stream>>>(a);

    // inverse DFT + residual + silu: h_b = silu(Gi[2048x128] @ O_b[128xW] + res_b)
    a = {};
    a.A = Gi;   a.lda = 128; a.sAz = 0;
    a.B = obuf; a.ldb = W;   a.sBz = (long)2 * MODES * W;
    a.M = G; a.N = W; a.K = 128;
    a.res = res16; a.C = h16; a.ldc = W; a.sCz = (long)G * W;
    gemm_wmma<128,128,32,false,false,EPI_INV_SILU>
        <<<dim3(1, G / 128, B), 256, 0, stream>>>(a);
  }

  // MLP head: t = silu(h @ f1 + b1) ; out = t . f2 + b2
  k_cvt<<<dim3((W * W + 255) / 256), 256, 0, stream>>>(f1W, f116, (long)W * W);
  a = {};
  a.A = h16; a.lda = W; a.B = f116; a.ldb = W;
  a.M = (int)BG; a.N = W; a.K = W;
  a.bias = f1b; a.C = t16; a.ldc = W;
  gemm_wmma<128,128,32,false,false,EPI_MLP1>
      <<<dim3(1, (int)(BG / 128), 1), 256, 0, stream>>>(a);

  k_dot<<<dim3((int)(BG / 8)), 256, 0, stream>>>(t16, f2W, f2b, (float*)d_out, BG);
}
